// SE3PointNeighborsConvolution_66125316489679
// MI455X (gfx1250) — matrix operand order repaired and checked by reading, not compiled, not run on hardware
//
#include <hip/hip_runtime.h>

// SE(3) point-neighbor convolution for MI455X (gfx1250, wave32).
// B=4, N=4096, CIN=COUT=32, NB=10, NN=64, RADIUS=0.123
// Stage 3 is a per-16-point-tile GEMM D[32x16] = W2[32x320] x T[320x16]
// mapped onto v_wmma_f32_16x16x32_f16 (f16 in, f32 accumulate).

#define B_      4
#define N_      4096
#define CIN_    32
#define COUT_   32
#define NB_     10
#define NN_     64
#define RADIUS_ 0.123f

#define PTS  16            // points per block
#define KTOT (NB_ * CIN_)  // 320 = contraction length of the final GEMM
#define STR  328           // padded LDS row stride in halves (656B: 16B-aligned rows)

typedef _Float16 h16v __attribute__((ext_vector_type(16)));
typedef _Float16 h8v  __attribute__((ext_vector_type(8)));
typedef float    f8v  __attribute__((ext_vector_type(8)));

union Frag { h16v v; h8v h[2]; };

__global__ __launch_bounds__(256)
void se3_nbr_conv_kernel(const float* __restrict__ feat,   // [B, CIN, N]
                         const float* __restrict__ geom,   // [B, N, 3]
                         const float* __restrict__ W,      // [COUT, CIN, NB]
                         float* __restrict__ out)          // [B, COUT, N]
{
    __shared__ int   s_cnt[PTS];
    __shared__ int   s_idx [PTS][NN_];
    __shared__ float s_dist[PTS][NN_];
    // rbf[16][64][10] f32 (40KB); after stage T it is dead and reused as W2h (f16)
    __shared__ __align__(16) float    s_union[PTS * NN_ * NB_];
    __shared__ __align__(16) _Float16 s_T[PTS * STR];   // T tile, f16, padded rows

    const int tid  = threadIdx.x;
    const int lane = tid & 31;
    const int wv   = tid >> 5;                // wave id 0..7 (wave32)
    const int grp  = blockIdx.x;              // 0 .. B*N/16 - 1
    const int b    = grp / (N_ / PTS);
    const int n0   = (grp % (N_ / PTS)) * PTS;

    if (tid < PTS) s_cnt[tid] = 0;
    __syncthreads();

    // ---------- phase 1: within-radius neighbor lists (wave wv -> points 2wv, 2wv+1)
    {
        const int pA = wv * 2, pB = wv * 2 + 1;
        const float* gb = geom + (size_t)b * N_ * 3;
        const float qax = gb[(n0 + pA) * 3 + 0];
        const float qay = gb[(n0 + pA) * 3 + 1];
        const float qaz = gb[(n0 + pA) * 3 + 2];
        const float qbx = gb[(n0 + pB) * 3 + 0];
        const float qby = gb[(n0 + pB) * 3 + 1];
        const float qbz = gb[(n0 + pB) * 3 + 2];
        const float R2 = RADIUS_ * RADIUS_;
        for (int m = lane; m < N_; m += 32) {
            const float x = gb[m * 3 + 0];
            const float y = gb[m * 3 + 1];
            const float z = gb[m * 3 + 2];
            {
                const float dx = x - qax, dy = y - qay, dz = z - qaz;
                const float d2 = dx * dx + dy * dy + dz * dz;
                if (d2 < R2) {
                    const int pos = atomicAdd(&s_cnt[pA], 1);
                    if (pos < NN_) {
                        s_idx[pA][pos]  = m;
                        s_dist[pA][pos] = sqrtf(fmaxf(d2, 1e-12f));
                    }
                }
            }
            {
                const float dx = x - qbx, dy = y - qby, dz = z - qbz;
                const float d2 = dx * dx + dy * dy + dz * dz;
                if (d2 < R2) {
                    const int pos = atomicAdd(&s_cnt[pB], 1);
                    if (pos < NN_) {
                        s_idx[pB][pos]  = m;
                        s_dist[pB][pos] = sqrtf(fmaxf(d2, 1e-12f));
                    }
                }
            }
        }
    }
    __syncthreads();

    // ---------- phase 2a: gaussian RBF table  rbf[pt][k][r], mask folded in
    {
        float* s_rbf = s_union;
        const float gamma = (NB_ / RADIUS_) * (NB_ / RADIUS_);
        const float step  = RADIUS_ / (NB_ - 1);
        for (int p = tid; p < PTS * NN_; p += 256) {
            const int pt = p >> 6;
            const int k  = p & (NN_ - 1);
            const int c  = s_cnt[pt];
            const int used = (c < NN_) ? c : NN_;
            const float d = (k < used) ? s_dist[pt][k] : 0.0f;
            const float v = (k < used) ? 1.0f : 0.0f;
            #pragma unroll
            for (int r = 0; r < NB_; ++r) {
                const float t = d - (float)r * step;
                s_rbf[p * NB_ + r] = v * __expf(-gamma * t * t);
            }
        }
    }
    __syncthreads();

    // ---------- phase 2b: T[pt][r*32+i] = sum_k rbf[pt][k][r] * feat[b][i][idx[pt][k]]
    {
        const float* s_rbf = s_union;
        #pragma unroll
        for (int s = 0; s < 2; ++s) {
            const int p  = tid + 256 * s;     // 0..511
            const int pt = p >> 5;            // wave-uniform -> rbf reads broadcast
            const int i  = p & 31;
            const float* fb = feat + ((size_t)b * CIN_ + i) * N_;
            float acc[NB_];
            #pragma unroll
            for (int r = 0; r < NB_; ++r) acc[r] = 0.0f;
            const int c = s_cnt[pt];
            const int used = (c < NN_) ? c : NN_;
            for (int k = 0; k < used; ++k) {
                const float f = fb[s_idx[pt][k]];
                const float* rb = s_rbf + (pt * NN_ + k) * NB_;
                #pragma unroll
                for (int r = 0; r < NB_; ++r) acc[r] = fmaf(rb[r], f, acc[r]);
            }
            _Float16* trow = s_T + pt * STR;
            #pragma unroll
            for (int r = 0; r < NB_; ++r) trow[r * 32 + i] = (_Float16)acc[r];
        }
    }
    __syncthreads();

    // ---------- phase 2c: stage W2h[o][r*32+i] (f16) over the dead rbf region
    {
        _Float16* W2h = (_Float16*)s_union;
        for (int t = tid; t < COUT_ * CIN_ * NB_; t += 256) {
            const int o   = t / (CIN_ * NB_);
            const int rem = t - o * (CIN_ * NB_);
            const int i   = rem / NB_;
            const int r   = rem - i * NB_;
            W2h[o * STR + r * 32 + i] = (_Float16)W[t];   // W flat = o*320 + i*10 + r
        }
    }
    __syncthreads();

    // ---------- phase 3: D[32x16] = W2[32x320] x T[320x16] via v_wmma_f32_16x16x32_f16
    // Waves 0 and 1 each own one 16-row M-tile; EXEC is all-ones per wave.
    if (tid < 64) {
        const _Float16* W2h = (const _Float16*)s_union;
        const int mt   = wv;                       // M-tile 0/1
        const int mrow = mt * 16 + (lane & 15);    // A row (16-bit A 16x32 layout)
        const int kbA  = (lane >> 4) * 8;          // lanes 0-15: K 0-7/16-23; 16-31: 8-15/24-31
        const int col  = lane & 15;                // B column (16-bit B 32x16 layout)
        const int kbB  = (lane >> 4) * 8;          // lanes 0-15: K 0-7/16-23; 16-31: 8-15/24-31
        f8v acc = {};
        #pragma unroll
        for (int ks = 0; ks < KTOT / 32; ++ks) {   // 10 K-steps of 32
            Frag a, bf;
            const _Float16* ap = W2h + mrow * STR + ks * 32;
            a.h[0] = *(const h8v*)(ap + kbA);
            a.h[1] = *(const h8v*)(ap + 16 + kbA);
            const _Float16* bp = s_T + col * STR + ks * 32 + kbB;
            bf.h[0] = *(const h8v*)(bp);
        bf.h[1] = *(const h8v*)(bp + 16);
            acc = __builtin_amdgcn_wmma_f32_16x16x32_f16(
                      /*neg_a=*/false, a.v, /*neg_b=*/false, bf.v,
                      /*c_mod=*/(short)0, acc, /*reuse_a=*/false, /*reuse_b=*/false);
        }
        // f32 C/D 16x16 layout: VGPR g -> M = (lane>=16 ? 8 : 0) + g, N = lane&15
        const int nn = n0 + (lane & 15);
        const int ob = mt * 16 + (lane >> 4) * 8;
        #pragma unroll
        for (int g = 0; g < 8; ++g)
            out[((size_t)b * COUT_ + ob + g) * N_ + nn] = acc[g];
    }
}

extern "C" void kernel_launch(void* const* d_in, const int* in_sizes, int n_in,
                              void* d_out, int out_size, void* d_ws, size_t ws_size,
                              hipStream_t stream) {
    (void)in_sizes; (void)n_in; (void)out_size; (void)d_ws; (void)ws_size;
    const float* feat = (const float*)d_in[0];   // [B, CIN, N]
    const float* geom = (const float*)d_in[1];   // [B, N, 3]
    const float* W    = (const float*)d_in[2];   // [COUT, CIN, NB]
    float* out = (float*)d_out;                  // [B, COUT, N]
    dim3 grid(B_ * N_ / PTS);                    // 1024 blocks
    dim3 block(256);                             // 8 wave32 waves
    hipLaunchKernelGGL(se3_nbr_conv_kernel, grid, block, 0, stream,
                       feat, geom, W, out);
}
